// MoELayer_2327872274774
// MI455X (gfx1250) — compile-verified, matrix operand order checked by reading, not run on hardware
//
#include <hip/hip_runtime.h>

// MoE top-2 FFN for MI455X (gfx1250, wave32, WMMA).
// T=8192 tokens, H=1024, F=4096, E=8. Weights pre-converted to bf16 (128MB,
// L2-resident). M=32 token tiles halve L2 weight-fragment traffic vs M=16.
// x rows gathered into LDS with GLOBAL_LOAD_ASYNC_TO_LDS_B128 (ASYNCcnt).
// launch_bounds(256,1) -> full VGPR budget, no spills of the 16 v8f
// persistent accumulators.

#define H_DIM 1024
#define F_DIM 4096
#define E_NUM 8
#define T_TOK 8192
#define MT    32          // token rows per tile (2 WMMA row-blocks)
#define FC    256         // F-chunk
#define XS    (H_DIM + 8) // LDS stride (elems), 16B-aligned rows
#define HS    (FC + 8)

typedef __attribute__((ext_vector_type(16))) __bf16 v16bf;
typedef __attribute__((ext_vector_type(8)))  float  v8f;

__device__ __forceinline__ unsigned short f2bf(float f) {
  union { float f; unsigned int u; } v; v.f = f;
  unsigned int r = v.u + 0x7FFFu + ((v.u >> 16) & 1u);  // RNE
  return (unsigned short)(r >> 16);
}

__device__ __forceinline__ float silu(float v) {
  // v * rcp(1+exp(-v)); v_rcp_f32 is ~1ulp, result is rounded to bf16 anyway
  return v * __builtin_amdgcn_rcpf(1.f + __expf(-v));
}

// x f32 -> bf16, elementwise (vectorized by 4)
__global__ void cvt_x_kernel(const float* __restrict__ in,
                             unsigned short* __restrict__ out, int n4) {
  int i = blockIdx.x * 256 + threadIdx.x;
  if (i < n4) {
    float4 v = ((const float4*)in)[i];
    ushort4 o;
    o.x = f2bf(v.x); o.y = f2bf(v.y); o.z = f2bf(v.z); o.w = f2bf(v.w);
    ((ushort4*)out)[i] = o;
  }
}

// in: [batch][R][C] f32 -> out: [batch][C][R] bf16  (R,C multiples of 32)
__global__ void transpose_cvt_kernel(const float* __restrict__ in,
                                     unsigned short* __restrict__ out,
                                     int R, int C) {
  __shared__ float tile[32][33];
  int b  = blockIdx.z;
  int c0 = blockIdx.x * 32;
  int r0 = blockIdx.y * 32;
  const float* ib = in + (size_t)b * R * C;
  unsigned short* ob = out + (size_t)b * R * C;
  int tx = threadIdx.x, ty = threadIdx.y;  // 32 x 8
  #pragma unroll
  for (int i = 0; i < 32; i += 8)
    tile[ty + i][tx] = ib[(size_t)(r0 + ty + i) * C + (c0 + tx)];
  __syncthreads();
  #pragma unroll
  for (int i = 0; i < 32; i += 8)
    ob[(size_t)(c0 + ty + i) * R + (r0 + tx)] = f2bf(tile[tx][ty + i]);
}

// One wave per token: logits -> softmax -> top-2 -> append to expert lists.
__global__ void gate_route_kernel(const float* __restrict__ x,
                                  const float* __restrict__ gw,
                                  int* __restrict__ counts,
                                  int* __restrict__ tok_id,
                                  float* __restrict__ tok_wt) {
  __shared__ float s_gw[E_NUM * H_DIM];  // 32 KB
  int tid = threadIdx.x;
  for (int i = tid; i < E_NUM * H_DIM; i += 256) s_gw[i] = gw[i];
  __syncthreads();
  int wave = tid >> 5, lane = tid & 31;
  int t = blockIdx.x * 8 + wave;
  float acc[E_NUM];
  #pragma unroll
  for (int e = 0; e < E_NUM; ++e) acc[e] = 0.f;
  const float* xr = x + (size_t)t * H_DIM;
  for (int h = lane; h < H_DIM; h += 32) {
    float xv = xr[h];
    #pragma unroll
    for (int e = 0; e < E_NUM; ++e) acc[e] += xv * s_gw[e * H_DIM + h];
  }
  #pragma unroll
  for (int e = 0; e < E_NUM; ++e)
    #pragma unroll
    for (int m = 16; m >= 1; m >>= 1) acc[e] += __shfl_xor(acc[e], m, 32);
  if (lane == 0) {
    float mx = acc[0];
    #pragma unroll
    for (int e = 1; e < E_NUM; ++e) mx = fmaxf(mx, acc[e]);
    float p[E_NUM], s = 0.f;
    #pragma unroll
    for (int e = 0; e < E_NUM; ++e) { p[e] = __expf(acc[e] - mx); s += p[e]; }
    float inv = 1.f / s;
    int i0 = 0;
    #pragma unroll
    for (int e = 1; e < E_NUM; ++e) if (p[e] > p[i0]) i0 = e;
    int i1 = (i0 == 0) ? 1 : 0;
    #pragma unroll
    for (int e = 0; e < E_NUM; ++e) if (e != i0 && p[e] > p[i1]) i1 = e;
    int pos0 = atomicAdd(&counts[i0], 1);
    tok_id[i0 * T_TOK + pos0] = t * 2 + 0;   // k-slot in LSB
    tok_wt[i0 * T_TOK + pos0] = p[i0] * inv;
    int pos1 = atomicAdd(&counts[i1], 1);
    tok_id[i1 * T_TOK + pos1] = t * 2 + 1;
    tok_wt[i1 * T_TOK + pos1] = p[i1] * inv;
  }
}

// Fused expert FFN, M=32 tile: out2[slot][tok] = (silu(x@w1+b1)@w2 + b2)*wt
__global__ void __launch_bounds__(256, 1)
moe_ffn_kernel(const unsigned short* __restrict__ xbf, // [T][H] bf16
               const unsigned short* __restrict__ w1t, // [E][F][H] bf16
               const unsigned short* __restrict__ w2t, // [E][H][F] bf16
               const float* __restrict__ b1,
               const float* __restrict__ b2,
               const int* __restrict__ counts,
               const int* __restrict__ tok_id,
               const float* __restrict__ tok_wt,
               float* __restrict__ out2)               // [2][T][H]
{
  __shared__ __align__(16) unsigned short s_x[MT * XS]; // 66 KB
  __shared__ __align__(16) unsigned short s_h[MT * HS]; // 16.9 KB
  __shared__ int   s_tok[MT];
  __shared__ float s_rw[MT];

  int e = blockIdx.y;
  int cnt = counts[e];
  int row0 = blockIdx.x * MT;
  if (row0 >= cnt) return;
  int rowsValid = min(MT, cnt - row0);

  int tid = threadIdx.x;
  int wave = tid >> 5, lane = tid & 31;
  int ln = lane & 15;
  int half8 = (lane < 16) ? 0 : 8;

  if (tid < MT) {
    if (tid < rowsValid) {
      s_tok[tid] = tok_id[e * T_TOK + row0 + tid];
      s_rw[tid]  = tok_wt[e * T_TOK + row0 + tid];
    } else { s_tok[tid] = 0; s_rw[tid] = 0.f; }
  }
  __syncthreads();

  // async-gather 32 token rows (bf16, 2KB each) into LDS: 16B per lane-issue
  {
    unsigned lds_base = (unsigned)(size_t)(&s_x[0]);
    #pragma unroll
    for (int j = 0; j < 16; ++j) {
      int idx = j * 256 + tid;            // 32 rows * 128 chunks
      int r  = idx >> 7;
      int ck = idx & 127;
      int tok = s_tok[r] >> 1;
      unsigned long long ga =
          (unsigned long long)(size_t)(xbf + (size_t)tok * H_DIM + ck * 8);
      unsigned la = lds_base + (unsigned)(r * (XS * 2) + ck * 16);
      asm volatile("global_load_async_to_lds_b128 %0, %1, off"
                   :: "v"(la), "v"(ga) : "memory");
    }
    asm volatile("s_wait_asynccnt 0" ::: "memory");
  }
  __syncthreads();

  // persistent accumulators: wave owns H cols [wave*128,+128) x 32 rows
  v8f acc0[8], acc1[8];
  #pragma unroll
  for (int t = 0; t < 8; ++t) {
    acc0[t] = (v8f){0,0,0,0,0,0,0,0};
    acc1[t] = (v8f){0,0,0,0,0,0,0,0};
  }

  for (int fc = 0; fc < F_DIM; fc += FC) {
    // -------- GEMM1: wave owns F cols [fc+wave*32,+32), 32 rows
    v8f c00 = (v8f){0,0,0,0,0,0,0,0}, c01 = c00, c10 = c00, c11 = c00;
    int n0 = fc + wave * 32 + ln;
    const unsigned short* w1b0 = w1t + ((size_t)e * F_DIM + n0)      * H_DIM;
    const unsigned short* w1b1 = w1t + ((size_t)e * F_DIM + n0 + 16) * H_DIM;
    #pragma unroll 2
    for (int k0 = 0; k0 < H_DIM; k0 += 32) {
      union { v16bf v; uint4 u[2]; } A0, A1, B0, B1;
      const unsigned short* ap0 = &s_x[ln * XS + k0 + half8];
      const unsigned short* ap1 = &s_x[(16 + ln) * XS + k0 + half8];
      A0.u[0] = *(const uint4*)(ap0);
      A0.u[1] = *(const uint4*)(ap0 + 16);
      A1.u[0] = *(const uint4*)(ap1);
      A1.u[1] = *(const uint4*)(ap1 + 16);
      B0.u[0] = *(const uint4*)(w1b0 + k0 + half8);
      B0.u[1] = *(const uint4*)(w1b0 + k0 + half8 + 16);
      B1.u[0] = *(const uint4*)(w1b1 + k0 + half8);
      B1.u[1] = *(const uint4*)(w1b1 + k0 + half8 + 16);
      c00 = __builtin_amdgcn_wmma_f32_16x16x32_bf16(false, A0.v, false, B0.v, (short)0, c00, false, false);
      c01 = __builtin_amdgcn_wmma_f32_16x16x32_bf16(false, A0.v, false, B1.v, (short)0, c01, false, false);
      c10 = __builtin_amdgcn_wmma_f32_16x16x32_bf16(false, A1.v, false, B0.v, (short)0, c10, false, false);
      c11 = __builtin_amdgcn_wmma_f32_16x16x32_bf16(false, A1.v, false, B1.v, (short)0, c11, false, false);
    }
    float bb0 = b1[(size_t)e * F_DIM + n0];
    float bb1 = b1[(size_t)e * F_DIM + n0 + 16];
    #pragma unroll
    for (int r = 0; r < 8; ++r) {
      int rowA = r + half8, rowB = 16 + r + half8;
      s_h[rowA * HS + wave * 32 + ln]      = f2bf(silu(c00[r] + bb0));
      s_h[rowA * HS + wave * 32 + ln + 16] = f2bf(silu(c01[r] + bb1));
      s_h[rowB * HS + wave * 32 + ln]      = f2bf(silu(c10[r] + bb0));
      s_h[rowB * HS + wave * 32 + ln + 16] = f2bf(silu(c11[r] + bb1));
    }
    __syncthreads();

    // -------- GEMM2: acc += h_chunk @ w2t, wave owns H cols [wave*128,+128)
    #pragma unroll 2
    for (int k0 = 0; k0 < FC; k0 += 32) {
      union { v16bf v; uint4 u[2]; } A2a, A2b;
      const unsigned short* ap2a = &s_h[ln * HS + k0 + half8];
      const unsigned short* ap2b = &s_h[(16 + ln) * HS + k0 + half8];
      A2a.u[0] = *(const uint4*)(ap2a);
      A2a.u[1] = *(const uint4*)(ap2a + 16);
      A2b.u[0] = *(const uint4*)(ap2b);
      A2b.u[1] = *(const uint4*)(ap2b + 16);
      #pragma unroll
      for (int t = 0; t < 8; ++t) {
        union { v16bf v; uint4 u[2]; } B2;
        int n2 = wave * 128 + t * 16 + ln;
        const unsigned short* w2b =
            w2t + ((size_t)e * H_DIM + n2) * F_DIM + fc + k0 + half8;
        B2.u[0] = *(const uint4*)(w2b);
        B2.u[1] = *(const uint4*)(w2b + 16);
        acc0[t] = __builtin_amdgcn_wmma_f32_16x16x32_bf16(false, A2a.v, false, B2.v, (short)0, acc0[t], false, false);
        acc1[t] = __builtin_amdgcn_wmma_f32_16x16x32_bf16(false, A2b.v, false, B2.v, (short)0, acc1[t], false, false);
      }
    }
    __syncthreads();
  }

  // epilogue
  #pragma unroll
  for (int t = 0; t < 8; ++t) {
    int col = wave * 128 + t * 16 + ln;
    float bb = b2[(size_t)e * H_DIM + col];
    #pragma unroll
    for (int r = 0; r < 8; ++r) {
      {
        int row = r + half8;
        if (row < rowsValid) {
          int id = s_tok[row];
          out2[((size_t)(id & 1) * T_TOK + (id >> 1)) * H_DIM + col] =
              (acc0[t][r] + bb) * s_rw[row];
        }
      }
      {
        int row = 16 + r + half8;
        if (row < rowsValid) {
          int id = s_tok[row];
          out2[((size_t)(id & 1) * T_TOK + (id >> 1)) * H_DIM + col] =
              (acc1[t][r] + bb) * s_rw[row];
        }
      }
    }
  }
}

__global__ void combine_kernel(const float* __restrict__ out2,
                               float* __restrict__ out, int n) {
  int i = blockIdx.x * blockDim.x + threadIdx.x;
  if (i < n) out[i] = out2[i] + out2[(size_t)T_TOK * H_DIM + i];
}

extern "C" void kernel_launch(void* const* d_in, const int* in_sizes, int n_in,
                              void* d_out, int out_size, void* d_ws, size_t ws_size,
                              hipStream_t stream) {
  const float* x      = (const float*)d_in[0];
  const float* gate_w = (const float*)d_in[1];
  const float* w1     = (const float*)d_in[2];
  const float* b1     = (const float*)d_in[3];
  const float* w2     = (const float*)d_in[4];
  const float* b2     = (const float*)d_in[5];
  float* out = (float*)d_out;

  // workspace layout (~208 MB)
  char* ws = (char*)d_ws;
  size_t off = 0;
  unsigned short* w1t = (unsigned short*)(ws + off); off += (size_t)E_NUM * F_DIM * H_DIM * 2;
  unsigned short* w2t = (unsigned short*)(ws + off); off += (size_t)E_NUM * H_DIM * F_DIM * 2;
  unsigned short* xbf = (unsigned short*)(ws + off); off += (size_t)T_TOK * H_DIM * 2;
  float* out2         = (float*)(ws + off);          off += (size_t)2 * T_TOK * H_DIM * 4;
  int*   counts       = (int*)(ws + off);            off += 256;
  int*   tok_id       = (int*)(ws + off);            off += (size_t)E_NUM * T_TOK * 4;
  float* tok_wt       = (float*)(ws + off);          off += (size_t)E_NUM * T_TOK * 4;

  hipMemsetAsync(counts, 0, 256, stream);

  dim3 tb(32, 8);
  transpose_cvt_kernel<<<dim3(F_DIM / 32, H_DIM / 32, E_NUM), tb, 0, stream>>>(w1, w1t, H_DIM, F_DIM);
  transpose_cvt_kernel<<<dim3(H_DIM / 32, F_DIM / 32, E_NUM), tb, 0, stream>>>(w2, w2t, F_DIM, H_DIM);
  cvt_x_kernel<<<(T_TOK * H_DIM / 4 + 255) / 256, 256, 0, stream>>>(x, xbf, T_TOK * H_DIM / 4);

  gate_route_kernel<<<T_TOK / 8, 256, 0, stream>>>(x, gate_w, counts, tok_id, tok_wt);

  moe_ffn_kernel<<<dim3(T_TOK / MT, E_NUM), 256, 0, stream>>>(
      xbf, w1t, w2t, b1, b2, counts, tok_id, tok_wt, out2);

  combine_kernel<<<(T_TOK * H_DIM + 255) / 256, 256, 0, stream>>>(out2, out, T_TOK * H_DIM);
}
